// HierarchicalBlock3D_2259152798592
// MI455X (gfx1250) — matrix operand order repaired and checked by reading, not compiled, not run on hardware
//
#include <hip/hip_runtime.h>
#include <hip/hip_bf16.h>
#include <cstdint>

typedef __bf16 bf16_t;
typedef __attribute__((ext_vector_type(16))) __bf16 v16bf;
typedef __attribute__((ext_vector_type(8)))  __bf16 v8bf;
typedef __attribute__((ext_vector_type(8)))  float  v8f;

#define DEV __device__ __forceinline__

#define B_   4
#define D0_  40
#define D1_  40
#define D2_  20
#define C_   256
#define NT_  128000      /* B*D0*D1*D2 */
#define NC_  1280        /* coarse tokens per batch (8*8*20) */
#define H_   4
#define DH_  64

// ---------- CDNA5 async copy: global -> LDS, 16B per lane, ASYNCcnt tracked ----------
// ISA 10.2: for a flat pointer into LDS, addr[31:0] is the workgroup-relative LDS
// offset, which is exactly what the async instruction's LDS-address VGPR wants.
DEV void async_copy_b128(void* lds_dst, const void* gsrc){
  unsigned off = (unsigned)(unsigned long long)lds_dst;
  asm volatile("global_load_async_to_lds_b128 %0, %1, off"
               :: "v"(off), "v"(gsrc) : "memory");
}
DEV void async_wait0(){
  asm volatile("s_wait_asynccnt 0x0" ::: "memory");
}

// ---------- WMMA fragment helpers (CDNA5 16x16x32 bf16, wave32) ----------
// A (16x32, row-major source): lane m = lane%16; element e -> k = 16*half + e
DEV v16bf load_a_frag(const bf16_t* rowp, int k0){
  int half = (threadIdx.x >> 4) & 1;
  return *(const v16bf*)(rowp + k0 + 16*half);
}
// B (32x16, n-major source row): lane n = lane%16; element e -> k = 16*half + e
DEV v16bf load_b_frag(const bf16_t* nrow, int k0){
  int half = (threadIdx.x >> 4) & 1;
  return *(const v16bf*)(nrow + k0 + 16*half);
}
DEV v8f wmma_bf16(v16bf a, v16bf b, v8f c){
  return __builtin_amdgcn_wmma_f32_16x16x32_bf16(false, a, false, b, (short)0, c, false, false);
}
// reductions confined to 16-lane halves (rows don't cross halves in C layout)
DEV float redmax16(float v){
  #pragma unroll
  for (int m=1;m<16;m<<=1) v = fmaxf(v, __shfl_xor(v, m, 32));
  return v;
}
DEV float redsum16(float v){
  #pragma unroll
  for (int m=1;m<16;m<<=1) v += __shfl_xor(v, m, 32);
  return v;
}

// ---------- RMSNorm (optionally fused residual add): one block per token ----------
__global__ __launch_bounds__(256) void k_rmsnorm(
    const float* __restrict__ a, const float* __restrict__ w,
    const float* __restrict__ addx, float* __restrict__ y)
{
  size_t tok = blockIdx.x;
  int tid = threadIdx.x;
  float v = a[tok*C_ + tid];
  float ss = v*v;
  #pragma unroll
  for (int m=1;m<32;m<<=1) ss += __shfl_xor(ss, m, 32);
  __shared__ float wsum[8];
  if ((tid&31)==0) wsum[tid>>5] = ss;
  __syncthreads();
  float tot = 0.f;
  #pragma unroll
  for (int i=0;i<8;i++) tot += wsum[i];
  float inv = rsqrtf(tot*(1.0f/C_) + 1e-6f);
  float r = v*w[tid]*inv;
  if (addx) r += addx[tok*C_ + tid];
  y[tok*C_ + tid] = r;
}

// ---------- depthwise 3x3x3 conv, y = x + conv(x); optional bf16 side copy ----------
__global__ __launch_bounds__(256) void k_dwconv(
    const float* __restrict__ x, const float* __restrict__ w,
    float* __restrict__ yf, bf16_t* __restrict__ yb)
{
  size_t idx = (size_t)blockIdx.x*256 + threadIdx.x;
  if (idx >= (size_t)NT_*64) return;
  int c4 = (int)(idx & 63)*4;
  size_t vox = idx >> 6;
  int d2 = (int)(vox % D2_);
  int d1 = (int)((vox / D2_) % D1_);
  int d0 = (int)((vox / (D2_*D1_)) % D0_);
  int b  = (int)(vox / ((size_t)D2_*D1_*D0_));
  float ax=0.f, ay=0.f, az=0.f, aw=0.f;
  for (int kz=0;kz<3;kz++){
    int z = d0+kz-1; if (z<0||z>=D0_) continue;
    for (int ky=0;ky<3;ky++){
      int yy = d1+ky-1; if (yy<0||yy>=D1_) continue;
      for (int kx=0;kx<3;kx++){
        int xx = d2+kx-1; if (xx<0||xx>=D2_) continue;
        size_t t = (((size_t)b*D0_+z)*D1_+yy)*D2_+xx;
        float4 xv = *(const float4*)(x + t*C_ + c4);
        float4 wv = *(const float4*)(w + ((kz*3+ky)*3+kx)*C_ + c4);
        ax += xv.x*wv.x; ay += xv.y*wv.y; az += xv.z*wv.z; aw += xv.w*wv.w;
      }
    }
  }
  float4 xc = *(const float4*)(x + vox*C_ + c4);
  float4 r = make_float4(xc.x+ax, xc.y+ay, xc.z+az, xc.w+aw);
  *(float4*)(yf + vox*C_ + c4) = r;
  if (yb){
    bf16_t* p = yb + vox*C_ + c4;
    p[0]=(bf16_t)r.x; p[1]=(bf16_t)r.y; p[2]=(bf16_t)r.z; p[3]=(bf16_t)r.w;
  }
}

// ---------- weight prep: transpose to [N][K] and convert to bf16 ----------
__global__ __launch_bounds__(256) void k_wprep(
    const float* __restrict__ qkv_w, const float* __restrict__ proj_w,
    bf16_t* __restrict__ wqT, bf16_t* __restrict__ wpT)
{
  int idx = blockIdx.x*256 + threadIdx.x;
  if (idx < 768*256){
    int n = idx / 256, k = idx % 256;
    wqT[idx] = (bf16_t)qkv_w[k*768 + n];
  } else if (idx < 768*256 + 256*256){
    int i = idx - 768*256;
    int n = i / 256, k = i % 256;
    wpT[i] = (bf16_t)proj_w[k*256 + n];
  }
}

// ---------- GEMM: out[M][N] = A[M][256] * Bt[N][256]^T + bias (+ residual) ----------
// block = 256 thr (8 waves), tile M=128 x N=64; Bt n-rows in LDS via async copy;
// fragment double-buffering across the (fully unrolled) K loop.
__global__ __launch_bounds__(256) void k_gemm(
    const bf16_t* __restrict__ A, int lda,
    const bf16_t* __restrict__ Bt,
    const float* __restrict__ bias,
    const float* __restrict__ residual, int ldr,
    float* __restrict__ outf, bf16_t* __restrict__ outb, int ldo)
{
  __shared__ alignas(32) bf16_t bsh[64*256];
  int mb = blockIdx.x, nb = blockIdx.y;
  int tid = threadIdx.x;
  const bf16_t* bsrc = Bt + (size_t)nb*64*256;
  #pragma unroll
  for (int i = 0; i < 8; i++)
    async_copy_b128(bsh + tid*8 + i*2048, bsrc + tid*8 + i*2048);
  async_wait0();
  __syncthreads();

  int lane = tid & 31, wv = tid >> 5;
  int half = lane >> 4, l16 = lane & 15;
  int m0 = mb*128 + wv*16;
  const bf16_t* arow = A + (size_t)(m0 + l16)*lda;
  v8f acc[4];
  #pragma unroll
  for (int j=0;j<4;j++)
    #pragma unroll
    for (int r=0;r<8;r++) acc[j][r]=0.f;

  v16bf af = load_a_frag(arow, 0);
  v16bf bfr[4];
  #pragma unroll
  for (int j=0;j<4;j++) bfr[j] = load_b_frag(bsh + (j*16 + l16)*256, 0);

  #pragma unroll
  for (int ks=0; ks<256; ks+=32){
    v16bf afn; v16bf bfn[4];
    if (ks < 224){
      afn = load_a_frag(arow, ks+32);
      #pragma unroll
      for (int j=0;j<4;j++) bfn[j] = load_b_frag(bsh + (j*16 + l16)*256, ks+32);
    }
    #pragma unroll
    for (int j=0;j<4;j++) acc[j] = wmma_bf16(af, bfr[j], acc[j]);
    if (ks < 224){
      af = afn;
      #pragma unroll
      for (int j=0;j<4;j++) bfr[j] = bfn[j];
    }
  }

  #pragma unroll
  for (int j=0;j<4;j++){
    int n = nb*64 + j*16 + l16;
    float bv = bias[n];
    #pragma unroll
    for (int r=0;r<8;r++){
      size_t m = (size_t)m0 + r + 8*half;
      float v = acc[j][r] + bv;
      if (residual) v += residual[m*(size_t)ldr + n];
      if (outb) outb[m*(size_t)ldo + n] = (bf16_t)v;
      else      outf[m*(size_t)ldo + n] = v;
    }
  }
}

// ---------- 5x5 mean pool over dims (0,1): qkv[NT][768] -> qkvc[B*NC][768] ----------
__global__ __launch_bounds__(256) void k_pool(
    const bf16_t* __restrict__ qkv, bf16_t* __restrict__ qkvc)
{
  size_t idx = (size_t)blockIdx.x*256 + threadIdx.x;
  if (idx >= (size_t)B_*NC_*768) return;
  int ch = (int)(idx % 768);
  int t  = (int)(idx / 768);
  int b  = t / NC_;
  int ct = t % NC_;
  int d2 = ct % 20;
  int c1 = (ct/20) & 7;
  int c0 = ct/160;
  float acc = 0.f;
  for (int t0=0;t0<5;t0++)
    for (int t1=0;t1<5;t1++){
      size_t tok = (((size_t)b*D0_ + c0*5+t0)*D1_ + c1*5+t1)*D2_ + d2;
      acc += (float)qkv[tok*768 + ch];
    }
  qkvc[idx] = (bf16_t)(acc*0.04f);
}

// window/token -> global token index (partition order from reference _part)
DEV int fine_token(int wid, int m){
  int b  = wid >> 8;        // 256 windows per batch (8*8*4)
  int r  = wid & 255;
  int i0 = r >> 5;
  int i1 = (r >> 2) & 7;
  int i2 = r & 3;
  int t0 = m/25, tr = m - t0*25;
  int t1 = tr/5, t2 = tr - t1*5;
  int d0 = i0*5+t0, d1 = i1*5+t1, d2 = i2*5+t2;
  return ((b*D0_ + d0)*D1_ + d1)*D2_ + d2;
}

// ---------- fine window attention: one block per (window, head) ----------
// LDS: vsh[64][128] @0, psh[128][128] @16K (overlaps dead ksh), ksh[128][64] @32K
__global__ __launch_bounds__(256) void k_fine_attn(
    const bf16_t* __restrict__ qkv, float* __restrict__ fine)
{
  __shared__ alignas(32) char smem[49152];
  bf16_t* vsh = (bf16_t*)smem;
  bf16_t* psh = (bf16_t*)(smem + 16384);
  bf16_t* ksh = (bf16_t*)(smem + 32768);
  int wid = blockIdx.x, h = blockIdx.y;
  int cb = h*DH_;
  int tid = threadIdx.x, lane = tid&31, wv = tid>>5, half = lane>>4, l16 = lane&15;
  // cooperative load: K [key][64] via async copy, V^T [d][128] scalar transpose
  {
    int key = tid >> 1, c0 = (tid & 1)*32;
    if (key < 125){
      int tok = fine_token(wid, key);
      const bf16_t* ksrc = qkv + (size_t)tok*768 + 256 + cb + c0;
      #pragma unroll
      for (int i=0;i<32;i+=8) async_copy_b128(ksh + key*64 + c0 + i, ksrc + i);
      const bf16_t* vsrc = qkv + (size_t)tok*768 + 512 + cb + c0;
      #pragma unroll
      for (int i=0;i<32;i++) vsh[(c0+i)*128 + key] = vsrc[i];
    } else {
      for (int i=0;i<32;i++){ ksh[key*64 + c0 + i] = (bf16_t)0.f; vsh[(c0+i)*128 + key] = (bf16_t)0.f; }
    }
  }
  async_wait0();
  __syncthreads();
  int mq = wv*16 + l16;
  int mtok = fine_token(wid, mq < 125 ? mq : 124);
  const bf16_t* qrow = qkv + (size_t)mtok*768 + cb;
  v16bf qa0 = load_a_frag(qrow, 0);
  v16bf qa1 = load_a_frag(qrow, 32);
  v8f s[8];
  #pragma unroll
  for (int j=0;j<8;j++){
    const bf16_t* krow = ksh + (j*16+l16)*64;
    v16bf b0 = load_b_frag(krow, 0);
    v16bf b1 = load_b_frag(krow, 32);
    v8f c;
    #pragma unroll
    for (int r=0;r<8;r++) c[r]=0.f;
    c = wmma_bf16(qa0, b0, c);
    c = wmma_bf16(qa1, b1, c);
    s[j]=c;
  }
  __syncthreads();          // all waves done with ksh (psh overlaps it)
  float l[8];
  #pragma unroll
  for (int j=0;j<8;j++){
    bool maskn = (j==7) && (l16 >= 13);   // keys 125..127 -> -inf
    #pragma unroll
    for (int r=0;r<8;r++)
      s[j][r] = maskn ? -1e30f : s[j][r]*0.125f;
  }
  #pragma unroll
  for (int r=0;r<8;r++){
    float mx = -1e30f;
    #pragma unroll
    for (int j=0;j<8;j++) mx = fmaxf(mx, s[j][r]);
    mx = redmax16(mx);
    float sm = 0.f;
    #pragma unroll
    for (int j=0;j<8;j++){ float p = __expf(s[j][r]-mx); s[j][r]=p; sm += p; }
    l[r] = redsum16(sm);
  }
  #pragma unroll
  for (int j=0;j<8;j++)
    #pragma unroll
    for (int r=0;r<8;r++)
      psh[(wv*16 + r + 8*half)*128 + j*16 + l16] = (bf16_t)s[j][r];
  __syncthreads();
  v8f o[4];
  #pragma unroll
  for (int j=0;j<4;j++)
    #pragma unroll
    for (int r=0;r<8;r++) o[j][r]=0.f;
  const bf16_t* prow = psh + (wv*16 + l16)*128;
  #pragma unroll
  for (int ks=0; ks<128; ks+=32){
    v16bf a = load_a_frag(prow, ks);
    v16bf bb[4];
    #pragma unroll
    for (int j=0;j<4;j++) bb[j] = load_b_frag(vsh + (j*16+l16)*128, ks);
    #pragma unroll
    for (int j=0;j<4;j++) o[j] = wmma_bf16(a, bb[j], o[j]);
  }
  #pragma unroll
  for (int r=0;r<8;r++){
    int mloc = wv*16 + r + 8*half;
    if (mloc < 125){
      int tok = fine_token(wid, mloc);
      float inv = 1.f / l[r];
      #pragma unroll
      for (int j=0;j<4;j++)
        fine[(size_t)tok*C_ + cb + j*16 + l16] = o[j][r]*inv;
    }
  }
}

// ---------- coarse global attention (flash-style): block per (qtile, head, batch) ----------
__global__ __launch_bounds__(256) void k_coarse_attn(
    const bf16_t* __restrict__ qkvc, bf16_t* __restrict__ oc)
{
  __shared__ alignas(32) char smem[49152];
  bf16_t* vsh = (bf16_t*)smem;
  bf16_t* psh = (bf16_t*)(smem + 16384);
  bf16_t* ksh = (bf16_t*)(smem + 32768);
  int qt = blockIdx.x, h = blockIdx.y, b = blockIdx.z;
  int cb = h*DH_;
  int tid=threadIdx.x, lane=tid&31, wv=tid>>5, half=lane>>4, l16=lane&15;
  const bf16_t* base = qkvc + (size_t)b*NC_*768;
  int mq = qt*128 + wv*16 + l16;
  const bf16_t* qrow = base + (size_t)mq*768 + cb;
  v16bf qa0 = load_a_frag(qrow, 0), qa1 = load_a_frag(qrow, 32);
  float mi[8], li[8];
  v8f o[4];
  #pragma unroll
  for (int r=0;r<8;r++){ mi[r] = -3.0e38f; li[r]=0.f; }
  #pragma unroll
  for (int j=0;j<4;j++)
    #pragma unroll
    for (int r=0;r<8;r++) o[j][r]=0.f;
  for (int kt=0; kt<10; kt++){
    __syncthreads();
    {
      int key = tid>>1, c0=(tid&1)*32;
      const bf16_t* ksrc = base + (size_t)(kt*128+key)*768 + 256 + cb + c0;
      #pragma unroll
      for (int i=0;i<32;i+=8) async_copy_b128(ksh + key*64 + c0 + i, ksrc + i);
      const bf16_t* vsrc = base + (size_t)(kt*128+key)*768 + 512 + cb + c0;
      #pragma unroll
      for (int i=0;i<32;i++) vsh[(c0+i)*128 + key] = vsrc[i];
    }
    async_wait0();
    __syncthreads();
    v8f s[8];
    #pragma unroll
    for (int j=0;j<8;j++){
      const bf16_t* krow = ksh + (j*16+l16)*64;
      v16bf b0 = load_b_frag(krow, 0);
      v16bf b1 = load_b_frag(krow, 32);
      v8f c;
      #pragma unroll
      for (int r=0;r<8;r++) c[r]=0.f;
      c = wmma_bf16(qa0, b0, c);
      c = wmma_bf16(qa1, b1, c);
      s[j]=c;
    }
    __syncthreads();        // ksh free -> psh writes may overlap
    #pragma unroll
    for (int r=0;r<8;r++){
      float lmx = -3.0e38f;
      #pragma unroll
      for (int j=0;j<8;j++){ s[j][r]*=0.125f; lmx = fmaxf(lmx, s[j][r]); }
      lmx = redmax16(lmx);
      float mnew = fmaxf(mi[r], lmx);
      float corr = __expf(mi[r] - mnew);
      float sm = 0.f;
      #pragma unroll
      for (int j=0;j<8;j++){ float p = __expf(s[j][r]-mnew); s[j][r]=p; sm += p; }
      sm = redsum16(sm);
      li[r] = li[r]*corr + sm;
      mi[r] = mnew;
      #pragma unroll
      for (int j=0;j<4;j++) o[j][r] *= corr;
    }
    #pragma unroll
    for (int j=0;j<8;j++)
      #pragma unroll
      for (int r=0;r<8;r++)
        psh[(wv*16 + r + 8*half)*128 + j*16 + l16] = (bf16_t)s[j][r];
    __syncthreads();
    const bf16_t* prow = psh + (wv*16 + l16)*128;
    #pragma unroll
    for (int ks=0; ks<128; ks+=32){
      v16bf a = load_a_frag(prow, ks);
      v16bf bb[4];
      #pragma unroll
      for (int j=0;j<4;j++) bb[j] = load_b_frag(vsh + (j*16+l16)*128, ks);
      #pragma unroll
      for (int j=0;j<4;j++) o[j] = wmma_bf16(a, bb[j], o[j]);
    }
  }
  #pragma unroll
  for (int r=0;r<8;r++){
    float inv = 1.f/li[r];
    int row = qt*128 + wv*16 + r + 8*half;
    #pragma unroll
    for (int j=0;j<4;j++)
      oc[((size_t)b*NC_ + row)*C_ + cb + j*16 + l16] = (bf16_t)(o[j][r]*inv);
  }
}

// ---------- low-rank on coarse tokens: lroc = (oc @ down) @ up ----------
__global__ __launch_bounds__(256) void k_lowrank(
    const bf16_t* __restrict__ oc, const float* __restrict__ down,
    const float* __restrict__ up, float* __restrict__ lroc)
{
  __shared__ float row[256];
  __shared__ float tv[64];
  int tok = blockIdx.x;
  int tid = threadIdx.x;
  row[tid] = (float)oc[(size_t)tok*C_ + tid];
  __syncthreads();
  if (tid < 64){
    float a = 0.f;
    for (int c=0;c<256;c++) a += row[c]*down[c*64 + tid];
    tv[tid] = a;
  }
  __syncthreads();
  float a = 0.f;
  #pragma unroll 8
  for (int r=0;r<64;r++) a += tv[r]*up[r*256 + tid];
  lroc[(size_t)tok*C_ + tid] = a;
}

// ---------- merged = fine + upsample(lroc), emit bf16 for proj GEMM ----------
__global__ __launch_bounds__(256) void k_merge(
    const float* __restrict__ fine, const float* __restrict__ lroc,
    bf16_t* __restrict__ merged)
{
  size_t idx = (size_t)blockIdx.x*256 + threadIdx.x;
  if (idx >= (size_t)NT_*C_) return;
  int c = (int)(idx & 255);
  size_t tk = idx >> 8;
  int b = (int)(tk / 32000);
  int rem = (int)(tk % 32000);
  int d0 = rem / 800;
  int d1 = (rem / 20) % 40;
  int d2 = rem % 20;
  int ct = ((d0/5)*8 + (d1/5))*20 + d2;
  merged[idx] = (bf16_t)(fine[idx] + lroc[((size_t)b*NC_ + ct)*C_ + c]);
}

extern "C" void kernel_launch(void* const* d_in, const int* in_sizes, int n_in,
                              void* d_out, int out_size, void* d_ws, size_t ws_size,
                              hipStream_t stream)
{
  (void)in_sizes; (void)n_in; (void)out_size; (void)ws_size;
  const float* x       = (const float*)d_in[0];
  const float* n1w     = (const float*)d_in[1];
  const float* anw     = (const float*)d_in[2];
  const float* qkv_w   = (const float*)d_in[3];
  const float* qkv_b   = (const float*)d_in[4];
  const float* proj_w  = (const float*)d_in[5];
  const float* proj_b  = (const float*)d_in[6];
  const float* dw_pre  = (const float*)d_in[7];
  const float* dw_post = (const float*)d_in[8];
  const float* lr_down = (const float*)d_in[9];
  const float* lr_up   = (const float*)d_in[10];
  float* y = (float*)d_out;

  char* ws = (char*)d_ws;
  float*  bufA  = (float*)(ws);                         // [NT][256] f32 (xn -> fine -> proj out)
  float*  bufB  = (float*)(ws + 131072000ull);          // [NT][256] f32 (xin -> post-conv)
  bf16_t* xinb  = (bf16_t*)(ws + 262144000ull);         // [NT][256] bf16 (xin, reused as merged)
  bf16_t* qkvb  = (bf16_t*)(ws + 327680000ull);         // [NT][768] bf16
  bf16_t* qkvc  = (bf16_t*)(ws + 524288000ull);         // [B*NC][768] bf16
  bf16_t* ocb   = (bf16_t*)(ws + 532152320ull);         // [B*NC][256] bf16
  float*  lroc  = (float*)(ws + 534773760ull);          // [B*NC][256] f32
  bf16_t* wqT   = (bf16_t*)(ws + 540016640ull);         // [768][256] bf16
  bf16_t* wpT   = (bf16_t*)(ws + 540409856ull);         // [256][256] bf16

  // 1) xn = rmsnorm(x, norm1_w)
  k_rmsnorm<<<NT_, 256, 0, stream>>>(x, n1w, nullptr, bufA);
  // 2) xin = xn + dwconv(xn, dw_pre)  (f32 + bf16 copies)
  k_dwconv<<<(NT_*64 + 255)/256, 256, 0, stream>>>(bufA, dw_pre, bufB, xinb);
  // 3) bf16-transposed weights
  k_wprep<<<(768*256 + 256*256 + 255)/256, 256, 0, stream>>>(qkv_w, proj_w, wqT, wpT);
  // 4) qkv = xin @ qkv_w + qkv_b   (bf16 out)
  k_gemm<<<dim3(NT_/128, 12), 256, 0, stream>>>(xinb, 256, wqT, qkv_b, nullptr, 0, nullptr, qkvb, 768);
  // 5) coarse pooling
  k_pool<<<(B_*NC_*768 + 255)/256, 256, 0, stream>>>(qkvb, qkvc);
  // 6) fine windowed attention -> bufA (f32)
  k_fine_attn<<<dim3(1024, H_), 256, 0, stream>>>(qkvb, bufA);
  // 7) coarse global attention -> ocb (bf16)
  k_coarse_attn<<<dim3(NC_/128, H_, B_), 256, 0, stream>>>(qkvc, ocb);
  // 8) low-rank on coarse tokens
  k_lowrank<<<B_*NC_, 256, 0, stream>>>(ocb, lr_down, lr_up, lroc);
  // 9) merged = fine + upsample(lroc)  (bf16, reuses xin bf16 buffer)
  k_merge<<<(NT_*C_ + 255)/256, 256, 0, stream>>>(bufA, lroc, xinb);
  // 10) out = merged @ proj_w + proj_b + xin  -> bufA (f32)
  k_gemm<<<dim3(NT_/128, 4), 256, 0, stream>>>(xinb, 256, wpT, proj_b, bufB, 256, bufA, nullptr, 256);
  // 11) a2 = out + dwconv(out, dw_post) -> bufB
  k_dwconv<<<(NT_*64 + 255)/256, 256, 0, stream>>>(bufA, dw_post, bufB, nullptr);
  // 12) y = x + rmsnorm(a2, attn_norm_w)
  k_rmsnorm<<<NT_, 256, 0, stream>>>(bufB, anw, x, y);
}
